// EncoderLayer_73263552135579
// MI455X (gfx1250) — matrix operand order repaired and checked by reading, not compiled, not run on hardware
//
#include <hip/hip_runtime.h>
#include <hip/hip_bf16.h>
#include <cstdint>
#include <cstddef>

// ---------------------------------------------------------------------------
// MI455X (gfx1250) encoder layer: WMMA f16 GEMMs with fragment-packed global
// layouts; B panels staged into LDS by the Tensor Data Mover (double-buffered,
// s_wait_tensorcnt), falling back to synchronous staging if the TDM builtin
// is unavailable. Workspace demand: ~1.1 GB in d_ws.
// ---------------------------------------------------------------------------

typedef __attribute__((ext_vector_type(16))) _Float16 v16h;
typedef __attribute__((ext_vector_type(8)))  _Float16 v8h;
typedef __attribute__((ext_vector_type(8)))  float    v8f;
typedef __attribute__((ext_vector_type(4)))  unsigned int u32x4;
typedef __attribute__((ext_vector_type(8)))  int      i32x8;
typedef __attribute__((ext_vector_type(4)))  int      i32x4;

#if __has_builtin(__builtin_amdgcn_tensor_load_to_lds) && \
    __has_builtin(__builtin_amdgcn_s_wait_tensorcnt)
#define HAVE_TDM 1
#else
#define HAVE_TDM 0
#endif

// ---- packed-layout index helpers ------------------------------------------
// A fragment (ISA 7.12.2, f16 16x32): lane l = hi*16+lr holds row m=mt*16+lr,
// halves i=0..7 -> k = kt*32 + hi*8 + i ; i=8..15 -> k = kt*32 + 16 + hi*8 + (i-8).
// PA[((mt*(K/32)+kt)*32 + l)*16 + i]
__device__ __forceinline__ size_t packA_index(size_t m, size_t k, int K) {
  size_t mt = m >> 4; int lr = (int)m & 15;
  size_t kt = k >> 5; int kk = (int)k & 31;
  int hi = (kk >> 3) & 1;                 // 0-7:0, 8-15:1, 16-23:0, 24-31:1
  int i  = (kk & 7) | ((kk >> 4) << 3);   // + 8 when kk >= 16
  return ((mt * (size_t)(K >> 5) + kt) * 32 + (size_t)(hi * 16 + lr)) * 16 + i;
}
// B fragment: lane l holds col n=nt*16+lr, halves i=0..7 -> k = kt*32 + hi*8 + i ; i=8..15 -> k = kt*32 + 16 + hi*8 + (i-8).
// PB[((kt*(N/16)+nt)*32 + l)*16 + i]

#if HAVE_TDM
// One-row TDM copy of a contiguous panel of `halves` f16 from global to LDS.
__device__ __forceinline__ void tdm_copy_to_lds(const _Float16* gsrc,
                                                void* ldsDst, int halves) {
  unsigned long long ga = (unsigned long long)(size_t)gsrc;
  unsigned lds = (unsigned)(size_t)ldsDst;     // low 32 bits = LDS byte offset
  u32x4 g0;
  g0[0] = 1u;                                   // count=1, user descriptor
  g0[1] = lds;                                  // lds_addr (bytes)
  g0[2] = (unsigned)(ga & 0xFFFFFFFFu);         // global_addr[31:0]
  g0[3] = (unsigned)((ga >> 32) & 0x01FFFFFFu)  // global_addr[56:32]
          | (2u << 30);                         // type = 2 ("image")
  i32x8 g1;
  g1[0] = (int)(1u << 16);                      // data_size = 1 -> 2 bytes
  g1[1] = (int)((unsigned)(halves & 0xFFFF) << 16);          // tensor_dim0 lo16
  g1[2] = (int)(((unsigned)halves >> 16) & 0xFFFFu) | (1 << 16); // dim0 hi16 | tensor_dim1=1
  g1[3] = (int)((unsigned)(halves & 0xFFFF) << 16);          // tile_dim0
  g1[4] = 1;                                    // tile_dim1 = 1, tile_dim2 = 0
  g1[5] = halves;                               // tensor_dim0_stride lo32
  g1[6] = 0;
  g1[7] = 0;
  i32x4 z4; z4[0] = z4[1] = z4[2] = z4[3] = 0;
  i32x8 z8;
  #pragma unroll
  for (int q = 0; q < 8; ++q) z8[q] = 0;
  // 6-arg toolchain form: (g0, g1, g2, g3, g4, cpol)
  __builtin_amdgcn_tensor_load_to_lds(g0, g1, z4, z4, z8, 0);
}
#endif

// ---------------------------- packing kernels ------------------------------

__global__ void pack_b_kernel(const float* __restrict__ B,
                              _Float16* __restrict__ PB, int K, int N) {
  size_t tid = (size_t)blockIdx.x * blockDim.x + threadIdx.x; // K*N
  if (tid >= (size_t)K * N) return;
  int i = (int)(tid & 15);
  int l = (int)((tid >> 4) & 31);
  size_t rest = tid >> 9;
  int nt = (int)(rest % (size_t)(N >> 4));
  int kt = (int)(rest / (size_t)(N >> 4));
  int k = kt * 32 + ((i < 8) ? ((l >> 4) * 8 + i) : (16 + (l >> 4) * 8 + (i - 8)));
  int n = nt * 16 + (l & 15);
  PB[tid] = (_Float16)B[(size_t)k * N + n];
}

__global__ void pack_a_kernel(const float* __restrict__ A,
                              _Float16* __restrict__ PA, int M, int K) {
  size_t tid = (size_t)blockIdx.x * blockDim.x + threadIdx.x; // M*K
  if (tid >= (size_t)M * K) return;
  int i = (int)(tid & 15);
  int l = (int)((tid >> 4) & 31);
  size_t rest = tid >> 9;
  int kt = (int)(rest % (size_t)(K >> 5));
  size_t mt = rest / (size_t)(K >> 5);
  int hi = l >> 4, lr = l & 15;
  int koff = (i < 8) ? (hi * 8 + i) : (16 + hi * 8 + (i - 8));
  size_t m = mt * 16 + lr;
  PA[tid] = (_Float16)A[m * K + kt * 32 + koff];
}

// LayerNorm over D=512, one 256-thread block per row, packed-A f16 output.
__global__ __launch_bounds__(256)
void layernorm_packA_kernel(const float* __restrict__ x,
                            const float* __restrict__ g,
                            const float* __restrict__ b,
                            _Float16* __restrict__ out) {
  constexpr int D = 512;
  __shared__ float sm[256];
  const size_t row = blockIdx.x;
  const int t = threadIdx.x;
  const float* xr = x + row * D;
  float v0 = xr[t], v1 = xr[t + 256];
  sm[t] = v0 + v1;
  __syncthreads();
  for (int o = 128; o > 0; o >>= 1) { if (t < o) sm[t] += sm[t + o]; __syncthreads(); }
  float mean = sm[0] * (1.0f / D);
  __syncthreads();
  float d0 = v0 - mean, d1 = v1 - mean;
  sm[t] = d0 * d0 + d1 * d1;
  __syncthreads();
  for (int o = 128; o > 0; o >>= 1) { if (t < o) sm[t] += sm[t + o]; __syncthreads(); }
  float inv = rsqrtf(sm[0] * (1.0f / D) + 1e-5f);
  out[packA_index(row, t,       D)] = (_Float16)(d0 * inv * g[t]       + b[t]);
  out[packA_index(row, t + 256, D)] = (_Float16)(d1 * inv * g[t + 256] + b[t + 256]);
}

// ------------------------------- WMMA GEMM ---------------------------------
// C[M,N] = A @ B from packed fragment layouts.
// Block = 256 threads (8 waves): wave w -> m-tile blockIdx.x*8+w,
// n-panel = blockIdx.y*128 (8 WMMA tiles). B panel (8 KB, contiguous in PB)
// staged per block in double-buffered LDS by the TDM, one K-step ahead.
template<bool BIAS, bool RELU, bool HALF_PACK, bool RES>
__global__ __launch_bounds__(256)
void gemm_wmma_packed(const _Float16* __restrict__ PA,
                      const _Float16* __restrict__ PB,
                      const float* __restrict__ bias,
                      const float* __restrict__ res,
                      float* __restrict__ Cf,
                      _Float16* __restrict__ Ch,
                      int M, int N, int K) {
  constexpr int NT = 8;                        // n-tiles per block
  constexpr int PANEL = NT * 32 * 16;          // 4096 halves = 8 KB
  __shared__ _Float16 Bsh[2][PANEL];
  const int t = threadIdx.x;
  const int wave = t >> 5, lane = t & 31;
  const int lr = lane & 15, hi = lane >> 4;
  const size_t mt = (size_t)blockIdx.x * 8 + wave;
  const size_t nb = blockIdx.y;                // panel of 8 n-tiles
  const int KT = K >> 5;
  const size_t NTILES = (size_t)(N >> 4);

  v8f acc[NT];
  #pragma unroll
  for (int j = 0; j < NT; ++j)
    #pragma unroll
    for (int r = 0; r < 8; ++r) acc[j][r] = 0.0f;

  const _Float16* aBase = PA + mt * (size_t)KT * 512 + (size_t)lane * 16;
  const _Float16* bPanel = PB + nb * (size_t)NT * 512;   // + kt*NTILES*512

#if HAVE_TDM
  if (wave == 0)
    tdm_copy_to_lds(bPanel, &Bsh[0][0], PANEL);          // prime kt=0
#endif

  for (int kt = 0; kt < KT; ++kt) {
    const int cur = kt & 1;
#if HAVE_TDM
    if (wave == 0) {
      if (kt + 1 < KT) {
        tdm_copy_to_lds(bPanel + (size_t)(kt + 1) * NTILES * 512,
                        &Bsh[cur ^ 1][0], PANEL);
        __builtin_amdgcn_s_wait_tensorcnt(1);  // kt's panel landed
      } else {
        __builtin_amdgcn_s_wait_tensorcnt(0);
      }
    }
    if (kt + 1 < KT)
      __builtin_prefetch(aBase + (size_t)(kt + 1) * 512, 0, 1);
    __syncthreads();
#else
    {
      const _Float16* src = bPanel + (size_t)kt * NTILES * 512;
      *(v8h*)&Bsh[cur][t * 8]        = *(const v8h*)(src + (size_t)t * 8);
      *(v8h*)&Bsh[cur][2048 + t * 8] = *(const v8h*)(src + 2048 + (size_t)t * 8);
      if (kt + 1 < KT) {
        __builtin_prefetch(src + NTILES * 512 + (size_t)t * 8, 0, 1);
        __builtin_prefetch(aBase + (size_t)(kt + 1) * 512, 0, 1);
      }
    }
    __syncthreads();
#endif

    const v16h a = *(const v16h*)(aBase + (size_t)kt * 512);
    #pragma unroll
    for (int j = 0; j < NT; ++j) {
      const v16h bf = *(const v16h*)&Bsh[cur][(j * 32 + lane) * 16];
      acc[j] = __builtin_amdgcn_wmma_f32_16x16x32_f16(
          false, a, false, bf, (short)0, acc[j], false, false);
    }
    __syncthreads();   // all waves done reading Bsh[cur] before it is refilled
  }

  // ---- epilogue (C layout: M = hi*8 + r, N = lane&15) ----
  #pragma unroll
  for (int j = 0; j < NT; ++j) {
    const size_t gn = nb * 128 + j * 16 + lr;
    const float bv = BIAS ? bias[gn] : 0.0f;
    #pragma unroll
    for (int r = 0; r < 8; ++r) {
      const size_t gm = mt * 16 + hi * 8 + r;
      float val = acc[j][r] + bv;
      if (RELU) val = fmaxf(val, 0.0f);
      if (RES)  val += res[gm * N + gn];
      if (HALF_PACK) Ch[packA_index(gm, gn, N)] = (_Float16)val; // packed for next GEMM
      else           Cf[gm * N + gn] = val;
    }
  }
}

// ------------------------- attention-side kernels --------------------------

// v[b,n,d] = max_j sigmoid(g2+pos_v_k)*(v2+pos_v_v), j over aa_idx (K2=16).
__global__ __launch_bounds__(256)
void compute_v_kernel(const float* __restrict__ qkvg,
                      const float* __restrict__ pos,
                      const int* __restrict__ aa_idx,
                      const unsigned char* __restrict__ aa_mask,
                      float* __restrict__ v) {
  constexpr int N = 2048;
  const size_t tid = (size_t)blockIdx.x * blockDim.x + threadIdx.x; // B*N*512
  const int d  = (int)(tid & 511);
  const size_t bn = tid >> 9;
  const int b  = (int)(bn >> 11);
  const int* ai = aa_idx + bn * 16;
  const unsigned char* am = aa_mask + bn * 16;
  const float* posr = pos + (bn * 32 + 16) * 1024;
  float m = -3.0e38f;
  #pragma unroll 4
  for (int j = 0; j < 16; ++j) {
    float v2 = 0.0f, g2 = 0.0f;
    if (!am[j]) {
      const float* row = qkvg + ((size_t)b * N + ai[j]) * 2048;
      v2 = row[1024 + d];
      g2 = row[1536 + d];
    }
    float pvk = posr[(size_t)j * 1024 + d];
    float pvv = posr[(size_t)j * 1024 + 512 + d];
    float sg  = 1.0f / (1.0f + __expf(-(g2 + pvk)));
    m = fmaxf(m, sg * (v2 + pvv));
  }
  v[tid] = m;
}

// Fused gather + QK^T + softmax (j=16) + AV. One block per (b,n),
// one wave32 per head (H=8, dk=64 -> 2 dk elems per lane). Packed-A output.
__global__ __launch_bounds__(256)
void attention_kernel(const float* __restrict__ qkvg,
                      const float* __restrict__ v,
                      const float* __restrict__ pos,
                      const int* __restrict__ idx,
                      _Float16* __restrict__ agg16) {
  constexpr int N = 2048;
  const size_t bn = blockIdx.x;
  const int b = (int)(bn >> 11);
  const int h = threadIdx.x >> 5;
  const int lane = threadIdx.x & 31;
  const int d0 = h * 64 + lane, d1 = d0 + 32;

  const float* qrow = qkvg + bn * 2048;
  const float q0 = qrow[d0], q1 = qrow[d1];
  const int* ir = idx + bn * 16;

  float s[16];
  float mx = -3.0e38f;
  #pragma unroll
  for (int j = 0; j < 16; ++j) {
    const size_t r = (size_t)b * N + ir[j];
    const float* krow = qkvg + r * 2048 + 512;
    const float* prow = pos + (bn * 32 + j) * 1024;
    float p = q0 * (krow[d0] + prow[d0]) + q1 * (krow[d1] + prow[d1]);
    #pragma unroll
    for (int o = 16; o > 0; o >>= 1) p += __shfl_xor(p, o, 32);
    p *= 0.125f;                       // dk^-0.5, dk=64
    s[j] = p;
    mx = fmaxf(mx, p);
  }
  float sum = 0.0f;
  #pragma unroll
  for (int j = 0; j < 16; ++j) { s[j] = __expf(s[j] - mx); sum += s[j]; }
  const float inv = 1.0f / sum;

  float a0 = 0.0f, a1 = 0.0f;
  #pragma unroll
  for (int j = 0; j < 16; ++j) {
    const size_t r = (size_t)b * N + ir[j];
    const float* vrow = v + r * 512;
    const float* prow = pos + (bn * 32 + j) * 1024 + 512;
    const float wj = s[j] * inv;
    a0 += wj * (vrow[d0] + prow[d0]);
    a1 += wj * (vrow[d1] + prow[d1]);
  }
  agg16[packA_index(bn, d0, 512)] = (_Float16)a0;
  agg16[packA_index(bn, d1, 512)] = (_Float16)a1;
}

// GLU: act[m,n] = relu(a)*sigmoid(g); a = ffn1[m,n], g = ffn1[m,2048+n].
// Packed-A output for the K=2048 ffn2 GEMM.
__global__ __launch_bounds__(256)
void glu_packA_kernel(const float* __restrict__ ffn1, _Float16* __restrict__ act) {
  const size_t i = (size_t)blockIdx.x * blockDim.x + threadIdx.x; // 4096*2048
  const size_t m = i >> 11, n = i & 2047;
  const float a = ffn1[m * 4096 + n];
  const float g = ffn1[m * 4096 + 2048 + n];
  act[packA_index(m, n, 2048)] =
      (_Float16)(fmaxf(a, 0.0f) * (1.0f / (1.0f + __expf(-g))));
}

// ------------------------------- launcher ----------------------------------

extern "C" void kernel_launch(void* const* d_in, const int* in_sizes, int n_in,
                              void* d_out, int out_size, void* d_ws, size_t ws_size,
                              hipStream_t stream) {
  (void)in_sizes; (void)n_in; (void)out_size; (void)ws_size;

  const float* x      = (const float*)d_in[0];
  const int*   idx    = (const int*)d_in[1];
  const float* dist   = (const float*)d_in[2];
  const int*   aa_idx = (const int*)d_in[4];
  const unsigned char* aa_mask = (const unsigned char*)d_in[5];
  const float* ln1_g  = (const float*)d_in[7];
  const float* ln1_b  = (const float*)d_in[8];
  const float* Wqkvg  = (const float*)d_in[9];
  const float* pos_w1 = (const float*)d_in[10];
  const float* pos_b1 = (const float*)d_in[11];
  const float* pos_w2 = (const float*)d_in[12];
  const float* pos_b2 = (const float*)d_in[13];
  const float* Wout   = (const float*)d_in[14];
  const float* ln2_g  = (const float*)d_in[15];
  const float* ln2_b  = (const float*)d_in[16];
  const float* ffn_w1 = (const float*)d_in[17];
  const float* ffn_b1 = (const float*)d_in[18];
  const float* ffn_w2 = (const float*)d_in[19];
  const float* ffn_b2 = (const float*)d_in[20];
  float* out = (float*)d_out;

  constexpr int B = 2, N = 2048, D = 512, J = 32;
  constexpr size_t BN  = (size_t)B * N;   // 4096 rows
  constexpr size_t BNJ = BN * J;          // 131072 rows

  // ---- workspace carve-up ----
  char* w = (char*)d_ws;
  size_t off = 0;
  auto take = [&](size_t bytes) -> char* {
    char* p = w + off;
    off += (bytes + 255) & ~(size_t)255;
    return p;
  };
  _Float16* h16p    = (_Float16*)take(BN * D * 2);              //   4 MB
  _Float16* wq_pb   = (_Float16*)take((size_t)D * 2048 * 2);    //   2 MB
  _Float16* pw1_pb  = (_Float16*)take((size_t)D * 1024 * 2);    //   1 MB
  _Float16* pw2_pb  = (_Float16*)take((size_t)1024 * 1024 * 2); //   2 MB
  _Float16* wout_pb = (_Float16*)take((size_t)D * D * 2);       // 0.5 MB
  _Float16* fw1_pb  = (_Float16*)take((size_t)D * 4096 * 2);    //   4 MB
  _Float16* fw2_pb  = (_Float16*)take((size_t)2048 * D * 2);    //   2 MB
  _Float16* distp   = (_Float16*)take(BNJ * D * 2);             // 128 MB
  _Float16* hid16p  = (_Float16*)take(BNJ * 1024 * 2);          // 256 MB
  float*    pos     = (float*)   take(BNJ * 1024 * 4);          // 512 MB
  float*    qkvg    = (float*)   take(BN * 2048 * 4);           //  32 MB
  float*    vbuf    = (float*)   take(BN * D * 4);              //   8 MB
  _Float16* agg16p  = (_Float16*)take(BN * D * 2);              //   4 MB
  float*    x1      = (float*)   take(BN * D * 4);              //   8 MB
  _Float16* h2p     = (_Float16*)take(BN * D * 2);              //   4 MB
  float*    ffn1    = (float*)   take(BN * 4096 * 4);           //  64 MB
  _Float16* actp    = (_Float16*)take(BN * 2048 * 2);           //  16 MB

  auto packB = [&](const float* src, _Float16* dst, int K, int Nn) {
    size_t n = (size_t)K * Nn;
    pack_b_kernel<<<(unsigned)((n + 255) / 256), 256, 0, stream>>>(src, dst, K, Nn);
  };

  // 1) weight packing (fragment-major) + dist pack (f32 -> packed f16)
  packB(Wqkvg,  wq_pb,   D,    2048);
  packB(pos_w1, pw1_pb,  D,    1024);
  packB(pos_w2, pw2_pb,  1024, 1024);
  packB(Wout,   wout_pb, D,    D);
  packB(ffn_w1, fw1_pb,  D,    4096);
  packB(ffn_w2, fw2_pb,  2048, D);
  pack_a_kernel<<<(unsigned)((BNJ * D + 255) / 256), 256, 0, stream>>>(
      dist, distp, (int)BNJ, D);

  // 2) LN1 -> packed f16
  layernorm_packA_kernel<<<(unsigned)BN, 256, 0, stream>>>(x, ln1_g, ln1_b, h16p);

  // 3) qkvg = h @ Wqkvg           (4096 x 2048 x 512)
  gemm_wmma_packed<false, false, false, false>
      <<<dim3(4096 / 128, 2048 / 128), 256, 0, stream>>>(
      h16p, wq_pb, nullptr, nullptr, qkvg, nullptr, 4096, 2048, 512);

  // 4) hidden = relu(dist @ pos_w1 + b1) -> packed f16 (131072 x 1024 x 512)
  gemm_wmma_packed<true, true, true, false>
      <<<dim3(131072 / 128, 1024 / 128), 256, 0, stream>>>(
      distp, pw1_pb, pos_b1, nullptr, nullptr, hid16p, 131072, 1024, 512);

  // 5) pos = hidden @ pos_w2 + b2                  (131072 x 1024 x 1024)
  gemm_wmma_packed<true, false, false, false>
      <<<dim3(131072 / 128, 1024 / 128), 256, 0, stream>>>(
      hid16p, pw2_pb, pos_b2, nullptr, pos, nullptr, 131072, 1024, 1024);

  // 6) v = max_j sigmoid(g2+pvk)*(v2+pvv)
  compute_v_kernel<<<(unsigned)(BN * 512 / 256), 256, 0, stream>>>(
      qkvg, pos, aa_idx, aa_mask, vbuf);

  // 7) fused attention -> packed agg
  attention_kernel<<<(unsigned)BN, 256, 0, stream>>>(qkvg, vbuf, pos, idx, agg16p);

  // 8) x1 = x + agg @ Wout                         (4096 x 512 x 512)
  gemm_wmma_packed<false, false, false, true>
      <<<dim3(4096 / 128, 512 / 128), 256, 0, stream>>>(
      agg16p, wout_pb, nullptr, x, x1, nullptr, 4096, 512, 512);

  // 9) LN2 -> packed f16
  layernorm_packA_kernel<<<(unsigned)BN, 256, 0, stream>>>(x1, ln2_g, ln2_b, h2p);

  // 10) ffn1 = h2 @ ffn_w1 + b1                    (4096 x 4096 x 512)
  gemm_wmma_packed<true, false, false, false>
      <<<dim3(4096 / 128, 4096 / 128), 256, 0, stream>>>(
      h2p, fw1_pb, ffn_b1, nullptr, ffn1, nullptr, 4096, 4096, 512);

  // 11) GLU -> packed act
  glu_packA_kernel<<<(unsigned)(BN * 2048 / 256), 256, 0, stream>>>(ffn1, actp);

  // 12) out = x1 + act @ ffn_w2 + b2               (4096 x 512 x 2048)
  gemm_wmma_packed<true, false, false, true>
      <<<dim3(4096 / 128, 512 / 128), 256, 0, stream>>>(
      actp, fw2_pb, ffn_b2, x1, out, nullptr, 4096, 512, 2048);
}
